// SingleHeadAttention_34600256537220
// MI455X (gfx1250) — matrix operand order repaired and checked by reading, not compiled, not run on hardware
//
#include <hip/hip_runtime.h>

// ---------------------------------------------------------------------------
// Single-head causal attention for MI455X (gfx1250, wave32, WMMA).
//   embedded: [4, 4096, 1024] f32   Wq/Wk/Wv: [1024, 1024] f32
//   out     : [4, 4096, 1024] f32
// Pipeline: f32->bf16 convert -> WMMA projection GEMMs (V stored transposed)
//           -> flash attention (32 q-rows per WG, 8 waves split the 1024 dim,
//              Q fragments hoisted into registers, K/V B-frags reused 2x,
//              global_prefetch of the next key block).
// ---------------------------------------------------------------------------

#define SEQ 4096
#define ED  1024
#define NB  4

typedef __attribute__((ext_vector_type(16))) __bf16 v16bf;
typedef __attribute__((ext_vector_type(8)))  __bf16 v8bf;
typedef __attribute__((ext_vector_type(4)))  __bf16 v4bf;
typedef __attribute__((ext_vector_type(8)))  float  v8f;

static __device__ inline v8f vzero8() {
  v8f z;
#pragma unroll
  for (int i = 0; i < 8; ++i) z[i] = 0.0f;
  return z;
}

// WMMA D = A(16x32 bf16) x B(32x16 bf16) + C(16x16 f32)
static __device__ inline v8f wmma_bf16(v16bf a, v16bf b, v8f c) {
  return __builtin_amdgcn_wmma_f32_16x16x32_bf16(
      /*neg_a=*/false, a, /*neg_b=*/false, b,
      /*c_mod=*/(short)0, c, /*reuse_a=*/false, /*reuse_b=*/false);
}

// A-fragment: lane holds row m = lane%16; K contiguous [16*hi,16*hi+16).
// rowptr points at (row, k0) of a row-major bf16 matrix; chunk 32B aligned.
static __device__ inline v16bf load_afrag(const __bf16* rowptr, int hi) {
  return *(const v16bf*)(rowptr + 16 * hi);
}

// ---------------------------------------------------------------------------
// Kernel 0: f32 -> bf16 conversion (4-wide).
// ---------------------------------------------------------------------------
__global__ __launch_bounds__(256) void cvt_f32_to_bf16(
    const float* __restrict__ in, __bf16* __restrict__ out, int n) {
  int i = (blockIdx.x * blockDim.x + threadIdx.x) * 4;
  if (i + 3 < n) {
    float4 v = *(const float4*)(in + i);
    v4bf o;
    o[0] = (__bf16)v.x; o[1] = (__bf16)v.y;
    o[2] = (__bf16)v.z; o[3] = (__bf16)v.w;
    *(v4bf*)(out + i) = o;
  }
}

// ---------------------------------------------------------------------------
// Kernel 1: projection GEMM.  Out[s,a] = sum_e X[s,e] * W[a,e]
// One wave per 16x16 tile; contraction 1024 = 32 WMMAs.
// transposed==0: store row-major [b][s][a] (Q, K)
// transposed==1: store [b][a][s]           (V^T, contiguous for P*V B-frags)
// ---------------------------------------------------------------------------
__global__ __launch_bounds__(256) void proj_gemm_bf16(
    const __bf16* __restrict__ X, const __bf16* __restrict__ W,
    __bf16* __restrict__ Out, int transposed) {
  const int lane = threadIdx.x & 31;
  const int wave = threadIdx.x >> 5;
  const int tile = blockIdx.x * 8 + wave;     // 0 .. 65535
  const int a_t  = tile & 63;                 // 64 a-tiles
  const int s_t  = (tile >> 6) & 255;         // 256 s-tiles
  const int b    = tile >> 14;                // 4 batches
  const int n  = lane & 15;
  const int hi = lane >> 4;

  const __bf16* xrow = X + (size_t)b * SEQ * ED + (size_t)(s_t * 16 + n) * ED;
  const __bf16* wrow = W + (size_t)(a_t * 16 + n) * ED;

  v8f c = vzero8();
#pragma unroll 4
  for (int k0 = 0; k0 < ED; k0 += 32) {
    v16bf a  = load_afrag(xrow + k0, hi);
    v16bf bb = *(const v16bf*)(wrow + k0 + 16 * hi);  // B-frag: 16 contig K
    c = wmma_bf16(a, bb, c);
  }

  if (!transposed) {
    // C elem r: (s = s_t*16 + r + 8*hi, a = a_t*16 + n); coalesced across lanes
    __bf16* o = Out + (size_t)b * SEQ * ED +
                (size_t)(s_t * 16 + 8 * hi) * ED + a_t * 16 + n;
#pragma unroll
    for (int r = 0; r < 8; ++r) o[(size_t)r * ED] = (__bf16)c[r];
  } else {
    v8bf pv;
#pragma unroll
    for (int r = 0; r < 8; ++r) pv[r] = (__bf16)c[r];
    // Vt[a][s]: 8 consecutive s per lane -> one 16B store
    __bf16* o = Out + (size_t)b * ED * SEQ +
                (size_t)(a_t * 16 + n) * SEQ + s_t * 16 + 8 * hi;
    *(v8bf*)o = pv;
  }
}

// ---------------------------------------------------------------------------
// Kernel 2: flash attention.
// Block = 256 threads = 8 waves; owns 32 query rows (2 subtiles of 16) of one
// batch.  Wave w handles output/contraction chunk a in [128w, 128w+128).
// Per 32-key block:
//   - partial 16x32 scores per wave & q-subtile (hoisted Q frags, K B-frags
//     reused for both subtiles) -> ds_add_f32 reduce in LDS
//   - online softmax (redundant per wave; 16-lane shfl_xor row reductions)
//   - P (bf16) staged in per-wave LDS -> P*V WMMAs; each V B-frag feeds both
//     q-subtiles' accumulators.
// ---------------------------------------------------------------------------
__global__ __launch_bounds__(256) void flash_attn_bf16(
    const __bf16* __restrict__ Q, const __bf16* __restrict__ K,
    const __bf16* __restrict__ Vt, float* __restrict__ Out) {
  __shared__ __align__(64) float  S_lds[2][16][32];        //  4 KB
  __shared__ __align__(64) __bf16 P_lds[8][2][16][32];     // 16 KB

  const int lane = threadIdx.x & 31;
  const int wave = threadIdx.x >> 5;
  const int n  = lane & 15;
  const int hi = lane >> 4;
  const int b  = blockIdx.y;
  const int q0 = blockIdx.x * 32;

  const size_t bo = (size_t)b * SEQ * ED;
  const __bf16* kbase  = K + bo;
  const __bf16* vtbase = Vt + (size_t)b * ED * SEQ;

  // ---- hoisted Q A-fragments: invariant over the whole key loop ----------
  v16bf qa[2][4];  // [q-subtile][k-chunk of the wave's 128-wide a-slice]
#pragma unroll
  for (int qt = 0; qt < 2; ++qt) {
    const __bf16* qrow =
        Q + bo + (size_t)(q0 + qt * 16 + n) * ED + wave * 128;
#pragma unroll
    for (int i = 0; i < 4; ++i) qa[qt][i] = load_afrag(qrow + i * 32, hi);
  }

  v8f acc[2][8];
#pragma unroll
  for (int qt = 0; qt < 2; ++qt)
#pragma unroll
    for (int t = 0; t < 8; ++t) acc[qt][t] = vzero8();
  float run_max[2][8], run_sum[2][8];
#pragma unroll
  for (int qt = 0; qt < 2; ++qt)
#pragma unroll
    for (int r = 0; r < 8; ++r) {
      run_max[qt][r] = -1.0e30f;
      run_sum[qt][r] = 0.0f;
    }

  const float scale = 0.03125f;  // 1/sqrt(1024)
  const int jmax = (q0 + 31) >> 5;

  for (int j = 0; j <= jmax; ++j) {
    const int kb = j * 32;

    // warm L2->WGP path for the next key block while this one computes
    if (j < jmax) {
      const int kb2 = kb + 32;
      __builtin_prefetch(kbase + (size_t)(kb2 + lane) * ED + wave * 128, 0, 0);
      __builtin_prefetch(
          vtbase + (size_t)(wave * 128 + lane * 4) * SEQ + kb2, 0, 0);
    }

    __syncthreads();  // prior-iteration S_lds reads complete
    {
      float* s = &S_lds[0][0][0];
      for (int i = threadIdx.x; i < 1024; i += 256) s[i] = 0.0f;
    }
    __syncthreads();

    // ---- partial scores over this wave's 128-wide a-chunk --------------
    v8f sa[2][2];  // [q-subtile][n-tile]
#pragma unroll
    for (int qt = 0; qt < 2; ++qt) { sa[qt][0] = vzero8(); sa[qt][1] = vzero8(); }
#pragma unroll
    for (int nt = 0; nt < 2; ++nt) {
      const __bf16* krow =
          kbase + (size_t)(kb + nt * 16 + n) * ED + wave * 128;
#pragma unroll
      for (int i = 0; i < 4; ++i) {
        v16bf bb = *(const v16bf*)(krow + i * 32 + 16 * hi);
        sa[0][nt] = wmma_bf16(qa[0][i], bb, sa[0][nt]);  // K frag reused
        sa[1][nt] = wmma_bf16(qa[1][i], bb, sa[1][nt]);
      }
    }
    // cross-wave reduction of partial scores
#pragma unroll
    for (int qt = 0; qt < 2; ++qt)
#pragma unroll
      for (int nt = 0; nt < 2; ++nt)
#pragma unroll
        for (int r = 0; r < 8; ++r)
          atomicAdd(&S_lds[qt][r + 8 * hi][nt * 16 + n], sa[qt][nt][r]);
    __syncthreads();

    // ---- scale + causal mask + online softmax (per q-subtile) ----------
#pragma unroll
    for (int qt = 0; qt < 2; ++qt) {
      float sv[2][8];
#pragma unroll
      for (int nt = 0; nt < 2; ++nt)
#pragma unroll
        for (int r = 0; r < 8; ++r) {
          float s = S_lds[qt][r + 8 * hi][nt * 16 + n] * scale;
          const int kg = kb + nt * 16 + n;
          const int qg = q0 + qt * 16 + r + 8 * hi;
          sv[nt][r] = (kg > qg) ? -1.0e30f : s;
        }

      float bm[8];
#pragma unroll
      for (int r = 0; r < 8; ++r) bm[r] = fmaxf(sv[0][r], sv[1][r]);
#pragma unroll
      for (int m = 1; m <= 8; m <<= 1)
#pragma unroll
        for (int r = 0; r < 8; ++r)
          bm[r] = fmaxf(bm[r], __shfl_xor(bm[r], m, 32));

      float nm[8], alpha[8], ps[8];
#pragma unroll
      for (int r = 0; r < 8; ++r) {
        nm[r] = fmaxf(run_max[qt][r], bm[r]);
        alpha[r] = __expf(run_max[qt][r] - nm[r]);
        ps[r] = 0.0f;
      }
#pragma unroll
      for (int nt = 0; nt < 2; ++nt)
#pragma unroll
        for (int r = 0; r < 8; ++r) {
          float p = __expf(sv[nt][r] - nm[r]);
          ps[r] += p;
          P_lds[wave][qt][r + 8 * hi][nt * 16 + n] = (__bf16)p;
        }
#pragma unroll
      for (int m = 1; m <= 8; m <<= 1)
#pragma unroll
        for (int r = 0; r < 8; ++r) ps[r] += __shfl_xor(ps[r], m, 32);
#pragma unroll
      for (int r = 0; r < 8; ++r) {
        run_sum[qt][r] = run_sum[qt][r] * alpha[r] + ps[r];
        run_max[qt][r] = nm[r];
      }
      // rescale running output accumulators for this subtile
#pragma unroll
      for (int t = 0; t < 8; ++t)
#pragma unroll
        for (int r = 0; r < 8; ++r) acc[qt][t][r] *= alpha[r];
    }

    // ---- P (16x32) x V-chunk (32x128); V B-frag reused for both subtiles
    v16bf pa0 = load_afrag(&P_lds[wave][0][n][0], hi);
    v16bf pa1 = load_afrag(&P_lds[wave][1][n][0], hi);
#pragma unroll
    for (int t = 0; t < 8; ++t) {
      const __bf16* vrow =
          vtbase + (size_t)(wave * 128 + t * 16 + n) * SEQ + kb + 16 * hi;
      v16bf bb = *(const v16bf*)vrow;
      acc[0][t] = wmma_bf16(pa0, bb, acc[0][t]);
      acc[1][t] = wmma_bf16(pa1, bb, acc[1][t]);
    }
  }

  // ---- epilogue: normalize, round to 1e-4, store f32 -------------------
#pragma unroll
  for (int qt = 0; qt < 2; ++qt) {
    float inv[8];
#pragma unroll
    for (int r = 0; r < 8; ++r) inv[r] = 1.0f / run_sum[qt][r];
#pragma unroll
    for (int t = 0; t < 8; ++t) {
      float* o = Out + bo + (size_t)(q0 + qt * 16 + 8 * hi) * ED +
                 wave * 128 + t * 16 + n;
#pragma unroll
      for (int r = 0; r < 8; ++r) {
        float v = acc[qt][t][r] * inv[r];
        o[(size_t)r * ED] = rintf(v * 10000.0f) * 1.0e-4f;
      }
    }
  }
}

// ---------------------------------------------------------------------------
// Host launcher.  d_in: {embedded, Wq, Wk, Wv} (f32).  d_out: f32 [4,4096,1024]
// d_ws layout (bytes):
//   Xb   @ 0          : 33554432
//   Wqb  @ 33554432   :  2097152
//   Wkb  @ 35651584   :  2097152
//   Wvb  @ 37748736   :  2097152
//   Qb   @ 39845888   : 33554432
//   Kb   @ 73400320   : 33554432
//   Vtb  @ 106954752  : 33554432   (total ~134 MB)
// ---------------------------------------------------------------------------
extern "C" void kernel_launch(void* const* d_in, const int* in_sizes, int n_in,
                              void* d_out, int out_size, void* d_ws,
                              size_t ws_size, hipStream_t stream) {
  const float* X  = (const float*)d_in[0];
  const float* Wq = (const float*)d_in[1];
  const float* Wk = (const float*)d_in[2];
  const float* Wv = (const float*)d_in[3];

  char* ws = (char*)d_ws;
  __bf16* Xb  = (__bf16*)(ws);
  __bf16* Wqb = (__bf16*)(ws + 33554432u);
  __bf16* Wkb = (__bf16*)(ws + 35651584u);
  __bf16* Wvb = (__bf16*)(ws + 37748736u);
  __bf16* Qb  = (__bf16*)(ws + 39845888u);
  __bf16* Kb  = (__bf16*)(ws + 73400320u);
  __bf16* Vtb = (__bf16*)(ws + 106954752u);

  const int nX = NB * SEQ * ED;   // 16,777,216
  const int nW = ED * ED;         //  1,048,576

  cvt_f32_to_bf16<<<nX / (256 * 4), 256, 0, stream>>>(X, Xb, nX);
  cvt_f32_to_bf16<<<nW / (256 * 4), 256, 0, stream>>>(Wq, Wqb, nW);
  cvt_f32_to_bf16<<<nW / (256 * 4), 256, 0, stream>>>(Wk, Wkb, nW);
  cvt_f32_to_bf16<<<nW / (256 * 4), 256, 0, stream>>>(Wv, Wvb, nW);

  // 65536 16x16 tiles per output, 8 waves (tiles) per block
  proj_gemm_bf16<<<8192, 256, 0, stream>>>(Xb, Wqb, Qb, 0);
  proj_gemm_bf16<<<8192, 256, 0, stream>>>(Xb, Wkb, Kb, 0);
  proj_gemm_bf16<<<8192, 256, 0, stream>>>(Xb, Wvb, Vtb, 1);

  flash_attn_bf16<<<dim3(SEQ / 32, NB), 256, 0, stream>>>(
      Qb, Kb, Vtb, (float*)d_out);
}
